// GCC_PHAT_15298673508501
// MI455X (gfx1250) — matrix operand order repaired and checked by reading, not compiled, not run on hardware
//
#include <hip/hip_runtime.h>
#include <hip/hip_bf16.h>
#include <math.h>

typedef __attribute__((ext_vector_type(16))) _Float16 v16h;
typedef __attribute__((ext_vector_type(8)))  float    v8f;

#define N_B     16
#define N_M     8
#define N_T     250
#define N_L     1024
#define KPAD    528            // 513 bins zero-padded for aligned float4 loads
#define N_ITER  33             // 1056 / 32 K-steps
#define N_PAIRS 28
#define N_LAGS  64
#define EPSF    1e-8f

// ---------------------------------------------------------------------------
// Kernel 0: B (twiddle) table in WMMA 32x16 f16 B-fragment layout:
// frag[(iter*4 + lagTile)*32 + lane][i]; h=lane/16: K=8h+i for i<8,
// K=16+8h+(i-8) for i>=8 (col=lane%16).
//   j even -> w*cos(2*pi*k*d/L)/L ; j odd -> -w*sin(2*pi*k*d/L)/L ; d = lag-32
// ---------------------------------------------------------------------------
__global__ __launch_bounds__(256)
void gcc_btable_kernel(_Float16* __restrict__ bt) {
    int idx = blockIdx.x * 256 + threadIdx.x;
    if (idx >= N_ITER * 4 * 32 * 16) return;
    int i       = idx & 15;
    int lane    = (idx >> 4) & 31;
    int lagTile = (idx >> 9) & 3;
    int iter    = idx >> 11;

    int klocal = ((i & 8) << 1) + ((lane >> 4) << 3) + (i & 7);
    int j      = iter * 32 + klocal;         // 0..1055
    int k      = j >> 1;                     // spectral bin
    int lag    = lagTile * 16 + (lane & 15); // 0..63
    float val  = 0.0f;
    if (k <= N_L / 2) {
        float w = (k == 0 || k == N_L / 2) ? 1.0f : 2.0f;
        int d   = lag - N_LAGS / 2;              // fftshift offset
        int m   = ((k * d) % N_L + N_L) % N_L;   // exact angle reduction
        float ang = 6.283185307179586f * (float)m / (float)N_L;
        float s, c;
        __sincosf(ang, &s, &c);
        val = ((j & 1) == 0) ? (w * c / (float)N_L) : (-w * s / (float)N_L);
    }
    bt[idx] = (_Float16)val;
}

// ---------------------------------------------------------------------------
// Kernel 1: 1024-pt radix-2 DIT complex FFT per row, one block per row.
// Twiddles from a 512-entry LDS table (one sincos per entry, then lookups).
// Writes bins 0..512 (+zero pad to 528) interleaved complex.
// ---------------------------------------------------------------------------
__global__ __launch_bounds__(256)
void gcc_fft_kernel(const float* __restrict__ x, float2* __restrict__ X) {
    __shared__ float2 z[N_L];
    __shared__ float2 tw[N_L / 2];
    const int row = blockIdx.x;              // (b*8+m)*250+t
    const int tid = threadIdx.x;
    const float* xr = x + (size_t)row * N_L;

    for (int p = tid; p < N_L / 2; p += 256) {
        float s, c;
        __sincosf(-6.283185307179586f * (float)p / (float)N_L, &s, &c);
        tw[p] = make_float2(c, s);
    }
    for (int n = tid; n < N_L; n += 256) {
        unsigned r = __brev((unsigned)n) >> 22;   // 10-bit reverse
        z[r] = make_float2(xr[n], 0.0f);
    }
    __syncthreads();

    int shift = 9;                                // pos*(1024/len) = pos<<shift
    for (int len = 2; len <= N_L; len <<= 1, --shift) {
        const int half = len >> 1;
        for (int bf = tid; bf < N_L / 2; bf += 256) {
            int grp = bf / half;
            int pos = bf - grp * half;
            int j0  = grp * len + pos;
            int j1  = j0 + half;
            float2 w2 = tw[pos << shift];
            float2 u = z[j0], w = z[j1];
            float tr = w2.x * w.x - w2.y * w.y;
            float ti = w2.x * w.y + w2.y * w.x;
            z[j1] = make_float2(u.x - tr, u.y - ti);
            z[j0] = make_float2(u.x + tr, u.y + ti);
        }
        __syncthreads();
    }

    float2* Xr = X + (size_t)row * KPAD;
    for (int k = tid; k < KPAD; k += 256)
        Xr[k] = (k <= N_L / 2) ? z[k] : make_float2(0.0f, 0.0f);
}

// ---------------------------------------------------------------------------
// Kernel 2: PHAT normalization + 64-lag inverse DFT as f16 WMMA GEMM.
// Phase 1: 128 threads cooperatively build ALL 33x32 A lane-fragments into
//          LDS (each fragment computed exactly once per block).
// Phase 2: 4 waves (one 16-lag tile each) run 11 x 3 independent WMMAs.
// ---------------------------------------------------------------------------
__global__ __launch_bounds__(128)
void gcc_phat_wmma_kernel(const float2* __restrict__ X,
                          const _Float16* __restrict__ bt,
                          float* __restrict__ out) {
    __shared__ alignas(32) _Float16 a_lds[N_ITER * 32 * 16];   // 33,792 B

    const int tTile = blockIdx.x;   // 0..15
    const int p     = blockIdx.y;   // 0..27
    const int b     = blockIdx.z;   // 0..15

    // decode triu pair (row-major: (0,1),(0,2)...,(1,2)...)
    int i1 = 0, rem = p, cnt = N_M - 1;
    while (rem >= cnt) { rem -= cnt; ++i1; --cnt; }
    int i2 = i1 + 1 + rem;

    const float2* xb1 = X + (size_t)(b * N_M + i1) * N_T * KPAD;
    const float2* xb2 = X + (size_t)(b * N_M + i2) * N_T * KPAD;

    // ---- Phase 1: stage A fragments (one per (iter,lane)) into LDS ----
    for (int w = threadIdx.x; w < N_ITER * 32; w += 128) {
        const int lane   = w & 31;
        const int iter   = w >> 5;
        const int laneHi = lane >> 4;
        const int rowm   = lane & 15;
        int t  = tTile * 16 + rowm;
        int tc = t < N_T ? t : N_T - 1;            // clamp; guarded at store

        const float2* x1 = xb1 + (size_t)tc * KPAD;
        const float2* x2 = xb2 + (size_t)tc * KPAD;
        const int kb = iter * 16 + laneHi * 4;     // lane's first bin group

        float4 a0 = *(const float4*)(x1 + kb);
        float4 a1 = *(const float4*)(x1 + kb + 2);
        float4 a2 = *(const float4*)(x1 + kb + 8);
        float4 a3 = *(const float4*)(x1 + kb + 10);
        float4 c0 = *(const float4*)(x2 + kb);
        float4 c1 = *(const float4*)(x2 + kb + 2);
        float4 c2 = *(const float4*)(x2 + kb + 8);
        float4 c3 = *(const float4*)(x2 + kb + 10);

        union { v16h v; _Float16 e[16]; } af;
        auto phat = [&](float r1, float m1, float r2, float m2, int o) {
            float rr  = r1 * r2 + m1 * m2;         // Re(X1*conj(X2))
            float ri  = m1 * r2 - r1 * m2;         // Im(X1*conj(X2))
            float inv = 1.0f / (sqrtf(rr * rr + ri * ri) + EPSF);
            af.e[o]     = (_Float16)(rr * inv);
            af.e[o + 1] = (_Float16)(ri * inv);
        };
        phat(a0.x, a0.y, c0.x, c0.y, 0);
        phat(a0.z, a0.w, c0.z, c0.w, 2);
        phat(a1.x, a1.y, c1.x, c1.y, 4);
        phat(a1.z, a1.w, c1.z, c1.w, 6);
        phat(a2.x, a2.y, c2.x, c2.y, 8);
        phat(a2.z, a2.w, c2.z, c2.w, 10);
        phat(a3.x, a3.y, c3.x, c3.y, 12);
        phat(a3.z, a3.w, c3.z, c3.w, 14);

        *(v16h*)(a_lds + (size_t)w * 16) = af.v;
    }
    __syncthreads();

    // ---- Phase 2: WMMA GEMM, 3 independent accumulators ----
    const int wave = threadIdx.x >> 5;   // lag tile
    const int lane = threadIdx.x & 31;

    v8f acc0 = (v8f){0.f, 0.f, 0.f, 0.f, 0.f, 0.f, 0.f, 0.f};
    v8f acc1 = acc0, acc2 = acc0;

    for (int iter = 0; iter < N_ITER; iter += 3) {
        v16h af0 = *(const v16h*)(a_lds + (size_t)((iter + 0) * 32 + lane) * 16);
        v16h af1 = *(const v16h*)(a_lds + (size_t)((iter + 1) * 32 + lane) * 16);
        v16h af2 = *(const v16h*)(a_lds + (size_t)((iter + 2) * 32 + lane) * 16);
        v16h bf0 = *(const v16h*)(bt + ((size_t)((iter + 0) * 4 + wave) * 32 + lane) * 16);
        v16h bf1 = *(const v16h*)(bt + ((size_t)((iter + 1) * 4 + wave) * 32 + lane) * 16);
        v16h bf2 = *(const v16h*)(bt + ((size_t)((iter + 2) * 4 + wave) * 32 + lane) * 16);

        acc0 = __builtin_amdgcn_wmma_f32_16x16x32_f16(false, af0, false, bf0,
                                                      (short)0, acc0, false, false);
        acc1 = __builtin_amdgcn_wmma_f32_16x16x32_f16(false, af1, false, bf1,
                                                      (short)0, acc1, false, false);
        acc2 = __builtin_amdgcn_wmma_f32_16x16x32_f16(false, af2, false, bf2,
                                                      (short)0, acc2, false, false);
    }

    union { v8f v; float e[8]; } acc;
    acc.v = acc0 + acc1 + acc2;

    // C layout: VGPR i -> M = i + 8*(lane/16), N = lane%16
    const int laneHi = lane >> 4;
    const int lag    = wave * 16 + (lane & 15);
    float* obase = out + (size_t)(b * N_PAIRS + p) * N_T * N_LAGS + lag;

    if (tTile * 16 + 15 < N_T) {                  // fast path: full tile
        #pragma unroll
        for (int i = 0; i < 8; ++i) {
            int tt = tTile * 16 + i + laneHi * 8;
            obase[(size_t)tt * N_LAGS] = acc.e[i];
        }
    } else {                                      // partial last tile
        #pragma unroll
        for (int i = 0; i < 8; ++i) {
            int tt = tTile * 16 + i + laneHi * 8;
            if (tt < N_T) obase[(size_t)tt * N_LAGS] = acc.e[i];
        }
    }
}

// ---------------------------------------------------------------------------
extern "C" void kernel_launch(void* const* d_in, const int* in_sizes, int n_in,
                              void* d_out, int out_size, void* d_ws, size_t ws_size,
                              hipStream_t stream) {
    (void)in_sizes; (void)n_in; (void)out_size; (void)ws_size;
    const float* x = (const float*)d_in[0];
    float*       o = (float*)d_out;

    float2* X = (float2*)d_ws;                                   // 135.17 MB
    const size_t xComplex = (size_t)N_B * N_M * N_T * KPAD;      // 16,896,000
    _Float16* bt = (_Float16*)((char*)d_ws + xComplex * sizeof(float2)); // +135 KB

    const int btElems = N_ITER * 4 * 32 * 16;                    // 67,584
    gcc_btable_kernel<<<(btElems + 255) / 256, 256, 0, stream>>>(bt);

    gcc_fft_kernel<<<N_B * N_M * N_T, 256, 0, stream>>>(x, X);

    dim3 g2((N_T + 15) / 16, N_PAIRS, N_B);                      // (16,28,16)
    gcc_phat_wmma_kernel<<<g2, 128, 0, stream>>>(X, bt, o);
}
